// PrimeExpLinear_62637803045435
// MI455X (gfx1250) — matrix operand order repaired and checked by reading, not compiled, not run on hardware
//
#include <hip/hip_runtime.h>
#include <hip/hip_bf16.h>

typedef __attribute__((ext_vector_type(16))) _Float16 v16h;
typedef __attribute__((ext_vector_type(8)))  _Float16 v8h;
typedef __attribute__((ext_vector_type(4)))  _Float16 v4h;
typedef __attribute__((ext_vector_type(8)))  float    v8f;

// ln of first 6 primes: 2,3,5,7,11,13
#define LP0 0.6931471805599453f
#define LP1 1.0986122886681098f
#define LP2 1.6094379124341003f
#define LP3 1.9459101090932196f
#define LP4 2.3978952727983707f
#define LP5 2.5649493574615367f
#define CLAMP_V 10.0f

// ---------------------------------------------------------------------------
// Phase 1a: W[o,k] = tanh(u[o,k]) * exp(clip(z[o,k,:]·logp))  ->  f16
// z_weight is (O, K, 6) contiguous: 24B per element, 8B aligned -> 3x float2
// ---------------------------------------------------------------------------
__global__ __launch_bounds__(256) void recon_w_kernel(
    const float* __restrict__ zw, const float* __restrict__ uw,
    _Float16* __restrict__ W16, int total) {
  int idx = blockIdx.x * blockDim.x + threadIdx.x;
  if (idx >= total) return;
  const float2* zp = reinterpret_cast<const float2*>(zw + (size_t)idx * 6);
  float2 a = zp[0], b = zp[1], c = zp[2];
  float lm = a.x * LP0 + a.y * LP1 + b.x * LP2 + b.y * LP3 + c.x * LP4 + c.y * LP5;
  lm = fminf(fmaxf(lm, -CLAMP_V), CLAMP_V);
  float w = tanhf(uw[idx]) * __expf(lm);
  W16[idx] = (_Float16)w;
}

// ---------------------------------------------------------------------------
// Phase 1b: bias[o] (kept in f32, added in the GEMM epilogue)
// ---------------------------------------------------------------------------
__global__ __launch_bounds__(256) void recon_bias_kernel(
    const float* __restrict__ zb, const float* __restrict__ ub,
    float* __restrict__ bias, int total) {
  int idx = blockIdx.x * blockDim.x + threadIdx.x;
  if (idx >= total) return;
  const float2* zp = reinterpret_cast<const float2*>(zb + (size_t)idx * 6);
  float2 a = zp[0], b = zp[1], c = zp[2];
  float lm = a.x * LP0 + a.y * LP1 + b.x * LP2 + b.y * LP3 + c.x * LP4 + c.y * LP5;
  lm = fminf(fmaxf(lm, -CLAMP_V), CLAMP_V);
  bias[idx] = tanhf(ub[idx]) * __expf(lm);
}

// ---------------------------------------------------------------------------
// Phase 1c: x f32 -> f16 (vectorized x4)
// ---------------------------------------------------------------------------
__global__ __launch_bounds__(256) void cvt_x_kernel(
    const float* __restrict__ x, _Float16* __restrict__ X16, int total4) {
  int idx = blockIdx.x * blockDim.x + threadIdx.x;
  if (idx >= total4) return;
  float4 v = reinterpret_cast<const float4*>(x)[idx];
  v4h h;
  h[0] = (_Float16)v.x; h[1] = (_Float16)v.y;
  h[2] = (_Float16)v.z; h[3] = (_Float16)v.w;
  reinterpret_cast<v4h*>(X16)[idx] = h;
}

// ---------------------------------------------------------------------------
// Phase 2: out = X16 @ W16.T + bias, WMMA f32_16x16x32_f16
// Block tile 128(M) x 128(N) x 64(K); 256 threads = 8 waves (wave32),
// wave grid 4(M) x 2(N): each wave -> 32x64 output = 2x4 fragments of 16x16.
// Two K=32 sub-steps per staged tile -> 16 WMMAs between barrier pairs.
// LDS rows padded to 72 halves (144B = 9x16B): all ds_load_b128 16B-aligned,
// 36-bank row stride -> conflict-free 16-row fragment gathers.
// ---------------------------------------------------------------------------
#define TILE_M 128
#define TILE_N 128
#define TILE_K 64
#define LDS_STRIDE 72

__global__ __launch_bounds__(256, 2) void gemm_wmma_kernel(
    const _Float16* __restrict__ X16, const _Float16* __restrict__ W16,
    const float* __restrict__ bias, float* __restrict__ out,
    int Ncols, int K) {
  __shared__ _Float16 sA[TILE_M * LDS_STRIDE];
  __shared__ _Float16 sB[TILE_N * LDS_STRIDE];

  const int tid  = threadIdx.x;
  const int lane = tid & 31;
  const int wave = tid >> 5;
  const int wm   = wave & 3;    // 0..3 -> M offset wm*32
  const int wn   = wave >> 2;   // 0..1 -> N offset wn*64
  const int blockM = blockIdx.x * TILE_M;
  const int blockN = blockIdx.y * TILE_N;

  const int lhalf = lane >> 4;  // 0/1: lane-group for fragment K mapping
  const int l16   = lane & 15;

  // cooperative-load geometry: 8-half (16B) chunks, 8 chunks per 64-half row.
  // chunk c = tid + 256*i (i=0..3): row = c>>3, kchunk = (c&7)*8.
  // Consecutive tids hit consecutive 16B segments -> fully coalesced.
  int arow[4], akc[4];
#pragma unroll
  for (int i = 0; i < 4; ++i) {
    int c = tid + 256 * i;
    arow[i] = c >> 3;
    akc[i]  = (c & 7) * 8;
  }

  const _Float16* Aptr = X16 + (size_t)blockM * K;
  const _Float16* Bptr = W16 + (size_t)blockN * K;

  const v8f zero = {0.f, 0.f, 0.f, 0.f, 0.f, 0.f, 0.f, 0.f};
  v8f acc[2][4] = {{zero, zero, zero, zero}, {zero, zero, zero, zero}};

  for (int k0 = 0; k0 < K; k0 += TILE_K) {
    // global -> regs (overlaps with previous iteration's MMA until barrier)
    v8h ra[4], rb[4];
#pragma unroll
    for (int i = 0; i < 4; ++i) {
      ra[i] = *reinterpret_cast<const v8h*>(Aptr + (size_t)arow[i] * K + k0 + akc[i]);
      rb[i] = *reinterpret_cast<const v8h*>(Bptr + (size_t)arow[i] * K + k0 + akc[i]);
    }
    if (k0 + TILE_K < K) {  // pull next K-tile toward the WGP (global_prefetch_b8)
      __builtin_prefetch(Aptr + (size_t)arow[0] * K + k0 + TILE_K + akc[0], 0, 3);
      __builtin_prefetch(Bptr + (size_t)arow[0] * K + k0 + TILE_K + akc[0], 0, 3);
    }
    __syncthreads();  // previous compute done reading LDS
#pragma unroll
    for (int i = 0; i < 4; ++i) {
      *reinterpret_cast<v8h*>(sA + arow[i] * LDS_STRIDE + akc[i]) = ra[i];
      *reinterpret_cast<v8h*>(sB + arow[i] * LDS_STRIDE + akc[i]) = rb[i];
    }
    __syncthreads();

#pragma unroll
    for (int kk = 0; kk < TILE_K; kk += 32) {
      // A fragment (16x32, ISA layout): lane<16 -> K {0..7,16..23}; lane>=16 -> +8
      v16h afrag[2];
#pragma unroll
      for (int i = 0; i < 2; ++i) {
        const _Float16* ap =
            sA + (wm * 32 + i * 16 + l16) * LDS_STRIDE + kk + lhalf * 8;
        v8h lo = *reinterpret_cast<const v8h*>(ap);
        v8h hi = *reinterpret_cast<const v8h*>(ap + 16);
        afrag[i] = __builtin_shufflevector(lo, hi, 0, 1, 2, 3, 4, 5, 6, 7,
                                           8, 9, 10, 11, 12, 13, 14, 15);
      }
      // B fragment (32x16): lane%16 = N column, lane<16 -> K {0..7,16..23}; lane>=16 -> +8
      v16h bfrag[4];
#pragma unroll
      for (int j = 0; j < 4; ++j) {
        const _Float16* bp =
            sB + (wn * 64 + j * 16 + l16) * LDS_STRIDE + kk + lhalf * 8;
        v8h lo = *reinterpret_cast<const v8h*>(bp);
        v8h hi = *reinterpret_cast<const v8h*>(bp + 16);
        bfrag[j] = __builtin_shufflevector(lo, hi, 0, 1, 2, 3, 4, 5, 6, 7,
                                           8, 9, 10, 11, 12, 13, 14, 15);
      }
#pragma unroll
      for (int i = 0; i < 2; ++i)
#pragma unroll
        for (int j = 0; j < 4; ++j)
          acc[i][j] = __builtin_amdgcn_wmma_f32_16x16x32_f16(
              false, afrag[i], false, bfrag[j], (short)0, acc[i][j], false, false);
    }
  }

  // Epilogue: C/D layout — element e is row M = e + 8*(lane>=16), col N = lane%16
#pragma unroll
  for (int j = 0; j < 4; ++j) {
    const int gcol = blockN + wn * 64 + j * 16 + l16;
    const float bv = bias[gcol];
#pragma unroll
    for (int i = 0; i < 2; ++i) {
      const int rbase = blockM + wm * 32 + i * 16 + lhalf * 8;
#pragma unroll
      for (int e = 0; e < 8; ++e)
        out[(size_t)(rbase + e) * Ncols + gcol] = acc[i][j][e] + bv;
    }
  }
}

// ---------------------------------------------------------------------------
extern "C" void kernel_launch(void* const* d_in, const int* in_sizes, int n_in,
                              void* d_out, int out_size, void* d_ws, size_t ws_size,
                              hipStream_t stream) {
  const float* x  = (const float*)d_in[0];  // (N, K)
  const float* zw = (const float*)d_in[1];  // (O, K, 6)
  const float* uw = (const float*)d_in[2];  // (O, K)
  const float* zb = (const float*)d_in[3];  // (O, 6)
  const float* ub = (const float*)d_in[4];  // (O,)
  float* out = (float*)d_out;

  const int O = in_sizes[4];            // 4096
  const int K = in_sizes[2] / O;        // 4096
  const int N = in_sizes[0] / K;        // 2048

  // workspace layout: W16 (O*K f16) | X16 (N*K f16) | bias (O f32)
  _Float16* W16 = (_Float16*)d_ws;
  _Float16* X16 = (_Float16*)((char*)d_ws + (size_t)O * K * sizeof(_Float16));
  float* bias   = (float*)((char*)d_ws + (size_t)O * K * sizeof(_Float16)
                                       + (size_t)N * K * sizeof(_Float16));

  const int totW = O * K;
  recon_w_kernel<<<(totW + 255) / 256, 256, 0, stream>>>(zw, uw, W16, totW);
  recon_bias_kernel<<<(O + 255) / 256, 256, 0, stream>>>(zb, ub, bias, O);
  const int totX4 = (N * K) / 4;
  cvt_x_kernel<<<(totX4 + 255) / 256, 256, 0, stream>>>(x, X16, totX4);

  dim3 grid(N / TILE_M, O / TILE_N);
  gemm_wmma_kernel<<<grid, 256, 0, stream>>>(X16, W16, bias, out, O, K);
}
